// CausalSelfAttention_84361747628629
// MI455X (gfx1250) — matrix operand order repaired and checked by reading, not compiled, not run on hardware
//
#include <hip/hip_runtime.h>
#include <hip/hip_bf16.h>
#include <cstdint>

// ---------------------------------------------------------------------------
// Causal self-attention forward for MI455X (gfx1250, wave32, WMMA).
// Roofline: ~140 GFLOP vs ~75 MB I/O -> matrix-unit bound. Use bf16 WMMA
// (v_wmma_f32_16x16x32_bf16) everywhere; flash-attention fusion avoids the
// 1.6 GB S-matrix round trip to HBM.
// ---------------------------------------------------------------------------

#define N_EMBD   768
#define N_HEADS  12
#define HEAD_DIM 64
#define BATCH    2
#define SEQ      4096

typedef __attribute__((ext_vector_type(16))) __bf16 bf16x16;
typedef __attribute__((ext_vector_type(8)))  __bf16 bf16x8;
typedef __attribute__((ext_vector_type(8)))  float  f32x8;

__device__ __forceinline__ unsigned short f32_to_bf16(float f) {
  union { float f; unsigned u; } v; v.f = f;
  unsigned u = v.u;
  u += 0x7FFFu + ((u >> 16) & 1u);          // round-to-nearest-even
  return (unsigned short)(u >> 16);
}

__device__ __forceinline__ bf16x16 combine8(bf16x8 lo, bf16x8 hi) {
  return __builtin_shufflevector(lo, hi, 0,1,2,3,4,5,6,7,8,9,10,11,12,13,14,15);
}

// ---------------------------------------------------------------------------
// f32 -> bf16 conversion
// ---------------------------------------------------------------------------
__global__ void f32_to_bf16_kernel(const float* __restrict__ src,
                                   unsigned short* __restrict__ dst, int n) {
  int i = blockIdx.x * blockDim.x + threadIdx.x;
  int stride = gridDim.x * blockDim.x;
  for (; i < n; i += stride) dst[i] = f32_to_bf16(src[i]);
}

// ---------------------------------------------------------------------------
// bf16 GEMM: C[M,N] = A[M,K] * B[K,N], A/B row-major bf16.
// Block tile 128x64, 256 threads = 8 waves (wave32), each wave does a 32x32
// sub-tile as 2x2 WMMA 16x16 accumulators, K stepped by 32 (WMMA bf16 K).
// MODE 0: f32 row-major output (final projection, writes d_out)
// MODE 1: bf16 output scattered into head-major Q/K/V layout [3][B][H][T][D]
// ---------------------------------------------------------------------------
template <int MODE>
__global__ __launch_bounds__(256)
void gemm_bf16_kernel(const unsigned short* __restrict__ A,
                      const unsigned short* __restrict__ Bm,
                      float* __restrict__ Cf,
                      unsigned short* __restrict__ Cq,
                      int M, int N, int K)
{
  __shared__ __align__(32) unsigned short As[128][32];  // A tile  [m][k]
  __shared__ __align__(32) unsigned short Bt[64][32];   // B tile transposed [n][k]

  const int tid    = threadIdx.x;
  const int wave   = tid >> 5;
  const int lane   = tid & 31;
  const int lane15 = lane & 15;
  const int half   = lane >> 4;         // lane 0-15 vs 16-31 (wave32)
  const int wm     = wave >> 1;         // 0..3
  const int wn     = wave & 1;          // 0..1
  const int m0     = blockIdx.y * 128;
  const int n0     = blockIdx.x * 64;

  f32x8 acc[2][2] = {};

  const int arow = tid >> 1, acs = (tid & 1) * 16;   // each thread: 16 bf16 of A
  const int brow = tid >> 3, bnc = (tid & 7) * 8;    // each thread: 8 bf16 of B

  for (int kk = 0; kk < K; kk += 32) {
    // ---- stage A tile (row-major) ----
    const uint4* ap = reinterpret_cast<const uint4*>(
        A + (size_t)(m0 + arow) * K + kk + acs);
    uint4 a0 = ap[0], a1 = ap[1];
    *reinterpret_cast<uint4*>(&As[arow][acs])     = a0;
    *reinterpret_cast<uint4*>(&As[arow][acs + 8]) = a1;

    // ---- stage B tile transposed: Bt[n][k] ----
    uint4 bv = *reinterpret_cast<const uint4*>(
        Bm + (size_t)(kk + brow) * N + n0 + bnc);
    alignas(16) unsigned short btmp[8];
    *reinterpret_cast<uint4*>(btmp) = bv;
    #pragma unroll
    for (int j = 0; j < 8; ++j) Bt[bnc + j][brow] = btmp[j];

    if (kk + 32 < K)   // gfx1250 global_prefetch_b8 on next A tile
      __builtin_prefetch(A + (size_t)(m0 + arow) * K + kk + 32 + acs, 0, 1);

    __syncthreads();

    #pragma unroll
    for (int sm = 0; sm < 2; ++sm) {
      // A frag 16x32: lane row = lane15, elems 0-7 = K[8*half..], 8-15 = K[16+8*half..]
      const int row = wm * 32 + sm * 16 + lane15;
      bf16x8 alo = *reinterpret_cast<const bf16x8*>(&As[row][8 * half]);
      bf16x8 ahi = *reinterpret_cast<const bf16x8*>(&As[row][16 + 8 * half]);
      bf16x16 afrag = combine8(alo, ahi);
      #pragma unroll
      for (int sn = 0; sn < 2; ++sn) {
        // B frag 32x16: lane col = lane15, elems 0-7 = K[8*half..], 8-15 = K[16+8*half..]
        const int col = wn * 32 + sn * 16 + lane15;
        bf16x16 bfrag = combine8(*reinterpret_cast<const bf16x8*>(&Bt[col][8 * half]),
                                 *reinterpret_cast<const bf16x8*>(&Bt[col][16 + 8 * half]));
        acc[sm][sn] = __builtin_amdgcn_wmma_f32_16x16x32_bf16(
            false, afrag, false, bfrag, (short)0, acc[sm][sn], false, false);
      }
    }
    __syncthreads();
  }

  // ---- store: C layout lane col = lane15, elem r -> row r + 8*half ----
  #pragma unroll
  for (int sm = 0; sm < 2; ++sm)
    #pragma unroll
    for (int sn = 0; sn < 2; ++sn)
      #pragma unroll
      for (int r = 0; r < 8; ++r) {
        int row = m0 + wm * 32 + sm * 16 + r + 8 * half;
        int col = n0 + wn * 32 + sn * 16 + lane15;
        float v = acc[sm][sn][r];
        if (MODE == 0) {
          Cf[(size_t)row * N + col] = v;
        } else {
          int bb  = row >> 12;            // row / SEQ
          int t   = row & (SEQ - 1);
          int mat = col / N_EMBD;         // 0=Q 1=K 2=V
          int rc  = col % N_EMBD;
          int h   = rc >> 6;
          int d   = rc & 63;
          size_t dst = ((size_t)mat * BATCH * N_HEADS
                        + (size_t)bb * N_HEADS + h) * SEQ * HEAD_DIM
                       + (size_t)t * HEAD_DIM + d;
          Cq[dst] = f32_to_bf16(v);
        }
      }
}

// ---------------------------------------------------------------------------
// Flash attention: one block per (b, h, 64-row q tile), 128 threads = 4 waves,
// each wave owns 16 q rows. Online softmax in f32, S/P stay on-chip.
// Q/K tiles row-major [t][d] in LDS (row-major K tile IS the Bt layout for
// Q*K^T). V transposed to [d][key] so it is the Bt layout for P*V.
// ---------------------------------------------------------------------------
__global__ __launch_bounds__(128)
void flash_attn_kernel(const unsigned short* __restrict__ Q,
                       const unsigned short* __restrict__ Km,
                       const unsigned short* __restrict__ V,
                       unsigned short* __restrict__ Y)
{
  __shared__ __align__(32) unsigned short Qs [64][64];   // [q][d]
  __shared__ __align__(32) unsigned short Ks [64][64];   // [key][d]
  __shared__ __align__(32) unsigned short Vts[64][64];   // [d][key]
  __shared__ __align__(32) unsigned short Ps [4][16][64];// per-wave P [q][key]

  const int tid    = threadIdx.x;
  const int wave   = tid >> 5;
  const int lane   = tid & 31;
  const int lane15 = lane & 15;
  const int half   = lane >> 4;
  const int qt = blockIdx.x, h = blockIdx.y, b = blockIdx.z;
  const int q0 = qt * 64;

  const size_t bh = ((size_t)b * N_HEADS + h) * SEQ * HEAD_DIM;
  const unsigned short* qp = Q  + bh;
  const unsigned short* kp = Km + bh;
  const unsigned short* vp = V  + bh;

  { // stage Q tile once
    const int row = tid >> 1, c0 = (tid & 1) * 32;
    const uint4* src = reinterpret_cast<const uint4*>(
        qp + (size_t)(q0 + row) * HEAD_DIM + c0);
    uint4* dst = reinterpret_cast<uint4*>(&Qs[row][c0]);
    #pragma unroll
    for (int i = 0; i < 4; ++i) dst[i] = src[i];
  }

  float mrow[8], lrow[8];
  f32x8 o[4] = {};
  #pragma unroll
  for (int r = 0; r < 8; ++r) { mrow[r] = -1e30f; lrow[r] = 0.f; }

  for (int kt = 0; kt <= qt; ++kt) {
    { // stage K (row-major) and V (transposed)
      const int row = tid >> 1, c0 = (tid & 1) * 32;
      const uint4* ksrc = reinterpret_cast<const uint4*>(
          kp + (size_t)(kt * 64 + row) * HEAD_DIM + c0);
      uint4* kdst = reinterpret_cast<uint4*>(&Ks[row][c0]);
      #pragma unroll
      for (int i = 0; i < 4; ++i) kdst[i] = ksrc[i];

      const uint4* vsrc = reinterpret_cast<const uint4*>(
          vp + (size_t)(kt * 64 + row) * HEAD_DIM + c0);
      alignas(16) unsigned short vt[32];
      #pragma unroll
      for (int i = 0; i < 4; ++i) *reinterpret_cast<uint4*>(vt + 8 * i) = vsrc[i];
      #pragma unroll
      for (int j = 0; j < 32; ++j) Vts[c0 + j][row] = vt[j];
    }
    __syncthreads();

    // ---- S = Q * K^T (16 q rows x 64 keys per wave) ----
    f32x8 s[4] = {};
    #pragma unroll
    for (int dstep = 0; dstep < 64; dstep += 32) {
      const int qrow = wave * 16 + lane15;
      bf16x8 qlo = *reinterpret_cast<const bf16x8*>(&Qs[qrow][dstep + 8 * half]);
      bf16x8 qhi = *reinterpret_cast<const bf16x8*>(&Qs[qrow][dstep + 16 + 8 * half]);
      bf16x16 afrag = combine8(qlo, qhi);
      #pragma unroll
      for (int j = 0; j < 4; ++j) {
        bf16x16 bfrag = combine8(
            *reinterpret_cast<const bf16x8*>(&Ks[16 * j + lane15][dstep + 8 * half]),
            *reinterpret_cast<const bf16x8*>(&Ks[16 * j + lane15][dstep + 16 + 8 * half]));
        s[j] = __builtin_amdgcn_wmma_f32_16x16x32_bf16(
            false, afrag, false, bfrag, (short)0, s[j], false, false);
      }
    }

    #pragma unroll
    for (int j = 0; j < 4; ++j)
      #pragma unroll
      for (int r = 0; r < 8; ++r) s[j][r] *= 0.125f;   // 1/sqrt(64)

    if (kt == qt) {  // causal mask on the diagonal tile
      #pragma unroll
      for (int j = 0; j < 4; ++j) {
        int keyg = kt * 64 + 16 * j + lane15;
        #pragma unroll
        for (int r = 0; r < 8; ++r) {
          int qg = q0 + wave * 16 + r + 8 * half;
          if (keyg > qg) s[j][r] = -1e30f;
        }
      }
    }

    // ---- online softmax (row reductions across the 16-lane half-wave) ----
    #pragma unroll
    for (int r = 0; r < 8; ++r) {
      float mx = s[0][r];
      #pragma unroll
      for (int j = 1; j < 4; ++j) mx = fmaxf(mx, s[j][r]);
      #pragma unroll
      for (int off = 8; off >= 1; off >>= 1)
        mx = fmaxf(mx, __shfl_xor(mx, off, 16));
      float mnew  = fmaxf(mrow[r], mx);
      float alpha = __expf(mrow[r] - mnew);
      float psum  = 0.f;
      #pragma unroll
      for (int j = 0; j < 4; ++j) {
        float p = __expf(s[j][r] - mnew);
        s[j][r] = p;
        psum += p;
      }
      #pragma unroll
      for (int off = 8; off >= 1; off >>= 1)
        psum += __shfl_xor(psum, off, 16);
      lrow[r] = lrow[r] * alpha + psum;
      mrow[r] = mnew;
      #pragma unroll
      for (int dt = 0; dt < 4; ++dt) o[dt][r] *= alpha;
    }

    // ---- P (C layout) -> LDS -> A-fragment layout ----
    #pragma unroll
    for (int j = 0; j < 4; ++j)
      #pragma unroll
      for (int r = 0; r < 8; ++r)
        Ps[wave][r + 8 * half][16 * j + lane15] = f32_to_bf16(s[j][r]);
    __syncthreads();

    // ---- O += P * V ----
    #pragma unroll
    for (int kstep = 0; kstep < 64; kstep += 32) {
      bf16x8 plo = *reinterpret_cast<const bf16x8*>(
          &Ps[wave][lane15][kstep + 8 * half]);
      bf16x8 phi = *reinterpret_cast<const bf16x8*>(
          &Ps[wave][lane15][kstep + 16 + 8 * half]);
      bf16x16 pfrag = combine8(plo, phi);
      #pragma unroll
      for (int dt = 0; dt < 4; ++dt) {
        bf16x16 vfrag = combine8(
            *reinterpret_cast<const bf16x8*>(&Vts[16 * dt + lane15][kstep + 8 * half]),
            *reinterpret_cast<const bf16x8*>(&Vts[16 * dt + lane15][kstep + 16 + 8 * half]));
        o[dt] = __builtin_amdgcn_wmma_f32_16x16x32_bf16(
            false, pfrag, false, vfrag, (short)0, o[dt], false, false);
      }
    }
    __syncthreads();
  }

  // ---- normalize and write y[b][t][h*64+d] (bf16, merged heads) ----
  #pragma unroll
  for (int r = 0; r < 8; ++r) {
    float inv = 1.f / lrow[r];
    int t = q0 + wave * 16 + r + 8 * half;
    #pragma unroll
    for (int dt = 0; dt < 4; ++dt) {
      int c = h * HEAD_DIM + 16 * dt + lane15;
      Y[((size_t)b * SEQ + t) * N_EMBD + c] = f32_to_bf16(o[dt][r] * inv);
    }
  }
}

// ---------------------------------------------------------------------------
// Host-side launch
// ---------------------------------------------------------------------------
extern "C" void kernel_launch(void* const* d_in, const int* in_sizes, int n_in,
                              void* d_out, int out_size, void* d_ws, size_t ws_size,
                              hipStream_t stream)
{
  (void)in_sizes; (void)n_in; (void)out_size; (void)ws_size;
  const float* x      = (const float*)d_in[0];
  const float* w_qkv  = (const float*)d_in[1];
  const float* w_proj = (const float*)d_in[2];
  float* out = (float*)d_out;

  const size_t M      = (size_t)BATCH * SEQ;             // 8192
  const size_t xN     = M * N_EMBD;                      // 6 291 456
  const size_t wqkvN  = (size_t)N_EMBD * 3 * N_EMBD;     // 1 769 472
  const size_t wprojN = (size_t)N_EMBD * N_EMBD;         //   589 824
  const size_t headN  = (size_t)BATCH * N_HEADS * SEQ * HEAD_DIM; // == xN

  unsigned short* ws      = (unsigned short*)d_ws;
  unsigned short* xbf     = ws;  ws += xN;
  unsigned short* wqkvbf  = ws;  ws += wqkvN;
  unsigned short* wprojbf = ws;  ws += wprojN;
  unsigned short* qkvbf   = ws;  ws += 3 * headN;   // Q | K | V head-major
  unsigned short* ybf     = ws;  ws += xN;

  f32_to_bf16_kernel<<<(int)((xN     + 255) / 256), 256, 0, stream>>>(x,      xbf,     (int)xN);
  f32_to_bf16_kernel<<<(int)((wqkvN  + 255) / 256), 256, 0, stream>>>(w_qkv,  wqkvbf,  (int)wqkvN);
  f32_to_bf16_kernel<<<(int)((wprojN + 255) / 256), 256, 0, stream>>>(w_proj, wprojbf, (int)wprojN);

  { // QKV projection: [8192 x 768] * [768 x 2304] -> head-major Q/K/V
    dim3 grid(3 * N_EMBD / 64, (unsigned)(M / 128));
    gemm_bf16_kernel<1><<<grid, 256, 0, stream>>>(
        xbf, wqkvbf, nullptr, qkvbf, (int)M, 3 * N_EMBD, N_EMBD);
  }
  { // fused causal flash attention
    dim3 grid(SEQ / 64, N_HEADS, BATCH);
    flash_attn_kernel<<<grid, 128, 0, stream>>>(
        qkvbf, qkvbf + headN, qkvbf + 2 * headN, ybf);
  }
  { // output projection: [8192 x 768] * [768 x 768] -> f32 out
    dim3 grid(N_EMBD / 64, (unsigned)(M / 128));
    gemm_bf16_kernel<0><<<grid, 256, 0, stream>>>(
        ybf, wprojbf, out, nullptr, (int)M, N_EMBD, N_EMBD);
  }
}
